// ReferenceMoeModel_36567351558810
// MI455X (gfx1250) — matrix operand order repaired and checked by reading, not compiled, not run on hardware
//
#include <hip/hip_runtime.h>
#include <stdint.h>

// Problem constants (B=2, S=4096, H=1024, I=512, E=8)
#define T_TOK 8192
#define H_DIM 1024
#define I_DIM 512
#define E_NUM 8

// Padded LDS row stride (elements) for 32-wide bf16 K tiles:
// 40 elems = 80 bytes -> 16B-aligned chunks for b128 traffic, and
// r*20 dwords mod 64 banks is distinct for r=0..15 (conflict-free frag reads).
#define SAP 40

typedef __attribute__((ext_vector_type(16))) __bf16 v16bf;
typedef __attribute__((ext_vector_type(8)))  float  v8f;

union FragAB { v16bf v; uint32_t u[8]; };

__device__ __forceinline__ unsigned short f2bf(float f) {
  union { float f; uint32_t u; } c; c.f = f;
  uint32_t u = c.u;
  u += 0x7FFFu + ((u >> 16) & 1u);   // round-to-nearest-even
  return (unsigned short)(u >> 16);
}

__device__ __forceinline__ float sigmoidf(float x) {
  return 1.0f / (1.0f + __expf(-x));
}

// CDNA5 async global->LDS copy, 16 bytes per lane, tracked by ASYNCcnt.
// LDS flat address low 32 bits == LDS byte offset (aperture truncation).
__device__ __forceinline__ void async_ld16(void* lptr, const void* gptr) {
  uint32_t l = (uint32_t)(uintptr_t)lptr;
  uint64_t g = (uint64_t)(uintptr_t)gptr;
  asm volatile("global_load_async_to_lds_b128 %0, %1, off"
               :: "v"(l), "v"(g) : "memory");
}
__device__ __forceinline__ void wait_async0() {
  asm volatile("s_wait_asynccnt 0x0" ::: "memory");
}

// ---------------------------------------------------------------- routing ----
__global__ void k_zero(int* __restrict__ cnt) {
  if (threadIdx.x < E_NUM) cnt[threadIdx.x] = 0;
}

// One wave (32 lanes) per token: 8 dot products of length H, shuffle-reduce.
__global__ void k_route(const float* __restrict__ x, const float* __restrict__ gw,
                        int* __restrict__ sel, float* __restrict__ rw,
                        int* __restrict__ cnt) {
  int wave = threadIdx.x >> 5;
  int lane = threadIdx.x & 31;
  int t = blockIdx.x * 8 + wave;
  const float* xr = x + (size_t)t * H_DIM;

  float acc[E_NUM];
#pragma unroll
  for (int e = 0; e < E_NUM; ++e) acc[e] = 0.0f;

  for (int h = lane; h < H_DIM; h += 32) {
    float xv = xr[h];
#pragma unroll
    for (int e = 0; e < E_NUM; ++e) acc[e] += xv * gw[e * H_DIM + h];
  }
#pragma unroll
  for (int e = 0; e < E_NUM; ++e) {
#pragma unroll
    for (int off = 16; off > 0; off >>= 1) acc[e] += __shfl_xor(acc[e], off, 32);
  }
  if (lane == 0) {
    int best = 0; float bv = acc[0];
#pragma unroll
    for (int e = 1; e < E_NUM; ++e) { if (acc[e] > bv) { bv = acc[e]; best = e; } }
    sel[t] = best;
    rw[t] = sigmoidf(bv);
    atomicAdd(&cnt[best], 1);
  }
}

__global__ void k_scan(const int* __restrict__ cnt, int* __restrict__ off,
                       int* __restrict__ cur) {
  if (threadIdx.x == 0) {
    int s = 0;
    for (int e = 0; e < E_NUM; ++e) { off[e] = s; cur[e] = s; s += cnt[e]; }
    off[E_NUM] = s;
  }
}

__global__ void k_assign(const int* __restrict__ sel, int* __restrict__ cur,
                         int* __restrict__ tok_of_slot) {
  int t = blockIdx.x * blockDim.x + threadIdx.x;
  if (t >= T_TOK) return;
  int e = sel[t];
  int slot = atomicAdd(&cur[e], 1);
  tok_of_slot[slot] = t;
}

// ------------------------------------------------------- bf16 conversions ----
__global__ void k_cvt_w(const float* __restrict__ wg, const float* __restrict__ wu,
                        const float* __restrict__ wd,
                        unsigned short* __restrict__ wgb,
                        unsigned short* __restrict__ wub,
                        unsigned short* __restrict__ wdb) {
  const size_t N = (size_t)E_NUM * I_DIM * H_DIM;
  size_t i = ((size_t)blockIdx.x * blockDim.x + threadIdx.x) * 4;
  const float* src; unsigned short* dst; size_t o;
  if (i < N)            { src = wg; dst = wgb; o = i; }
  else if (i < 2 * N)   { src = wu; dst = wub; o = i - N; }
  else                  { src = wd; dst = wdb; o = i - 2 * N; }
  float4 v = *(const float4*)(src + o);
  dst[o + 0] = f2bf(v.x);
  dst[o + 1] = f2bf(v.y);
  dst[o + 2] = f2bf(v.z);
  dst[o + 3] = f2bf(v.w);
}

// xg[slot] = bf16(x[tok] * rw[tok])  (input-side routing weight, as reference)
__global__ void k_fill_xg(const float* __restrict__ x, const float* __restrict__ rw,
                          const int* __restrict__ tok_of_slot,
                          unsigned short* __restrict__ xg) {
  int slot = blockIdx.x;
  int t = tok_of_slot[slot];
  float r = rw[t];
  const float* xr = x + (size_t)t * H_DIM;
  unsigned short* dr = xg + (size_t)slot * H_DIM;
  for (int h = threadIdx.x; h < H_DIM; h += blockDim.x) dr[h] = f2bf(xr[h] * r);
}

// -------------------------------------------------- WMMA fragment helpers ----
// A-matrix 16x32 bf16 (ISA layout): lane<16 -> M=lane, K={0..7,16..23};
// lane>=16 -> M=lane-16, K={8..15,24..31}; VGPR j holds K pair.
__device__ __forceinline__ int kofsA(int j, int hi) {
  return 2 * (j & 3) + ((j & 4) ? 16 : 0) + hi * 8;
}
// B-matrix 32x16 bf16: lanes 0-15 hold K={0..7,16..23} (N=lane), lanes 16-31 K={8..15,24..31}.
__device__ __forceinline__ int kofsB(int j, int hi) {
  return 2 * (j & 3) + ((j & 4) ? 16 : 0) + hi * 8;
}

__device__ __forceinline__ void loadFragA(FragAB& f, const unsigned short* sh,
                                          int row, int hi) {
#pragma unroll
  for (int j = 0; j < 8; ++j)
    f.u[j] = *(const uint32_t*)&sh[row * SAP + kofsA(j, hi)];
}
__device__ __forceinline__ void loadFragB(FragAB& f, const unsigned short* sh,
                                          int row, int hi) {
#pragma unroll
  for (int j = 0; j < 8; ++j)
    f.u[j] = *(const uint32_t*)&sh[row * SAP + kofsB(j, hi)];
}

// ------------------------------------------------ GEMM1: fused gate+up+silu --
// g = Xg @ Wg^T, u = Xg @ Wu^T, a = u * silu(g); per expert-token-segment.
// Block: 128 threads (4 waves), tile 64(M) x 32(N), K-step 32 over H.
// Double-buffered LDS filled with global_load_async_to_lds_b128.
__global__ __launch_bounds__(128) void k_gemm_gu(
    const unsigned short* __restrict__ xg,
    const unsigned short* __restrict__ wgb,
    const unsigned short* __restrict__ wub,
    const int* __restrict__ off,
    unsigned short* __restrict__ abuf) {
  int e = blockIdx.z;
  int base = off[e];
  int cnt = off[e + 1] - base;
  int m0 = blockIdx.x * 64;
  if (m0 >= cnt) return;
  int n0 = blockIdx.y * 32;

  __shared__ unsigned short shA[2][64 * SAP];
  __shared__ unsigned short shG[2][32 * SAP];
  __shared__ unsigned short shU[2][32 * SAP];

  int tid = threadIdx.x;
  int lane = tid & 31, wv = tid >> 5;
  int hi = lane >> 4, ln = lane & 15;

  const v8f vzero = {0.f, 0.f, 0.f, 0.f, 0.f, 0.f, 0.f, 0.f};
  v8f accG[2] = {vzero, vzero};
  v8f accU[2] = {vzero, vzero};

  const unsigned short* aG  = xg  + (size_t)base * H_DIM;
  const unsigned short* wgE = wgb + (size_t)e * I_DIM * H_DIM;
  const unsigned short* wuE = wub + (size_t)e * I_DIM * H_DIM;

  // Per-thread staging coordinates (branch-free; rows beyond cnt are garbage
  // but only influence output rows >= cnt, which are guarded at the store).
  int rA0 = (tid)        >> 2, cA0 = ((tid)        & 3) * 8;  // A round 0
  int rA1 = (tid + 128)  >> 2, cA1 = ((tid + 128)  & 3) * 8;  // A round 1
  int rB  = tid >> 2,          cB  = (tid & 3) * 8;           // B (32 rows)

  auto stage = [&](int k0, int b) {
    async_ld16(&shA[b][rA0 * SAP + cA0],
               aG + (size_t)(m0 + rA0) * H_DIM + k0 + cA0);
    async_ld16(&shA[b][rA1 * SAP + cA1],
               aG + (size_t)(m0 + rA1) * H_DIM + k0 + cA1);
    async_ld16(&shG[b][rB * SAP + cB],
               wgE + (size_t)(n0 + rB) * H_DIM + k0 + cB);
    async_ld16(&shU[b][rB * SAP + cB],
               wuE + (size_t)(n0 + rB) * H_DIM + k0 + cB);
  };

  const int NK = H_DIM / 32;
  stage(0, 0);
  for (int ks = 0; ks < NK; ++ks) {
    int b = ks & 1;
    wait_async0();       // this wave's stage(ks) copies have landed in LDS
    __syncthreads();     // all waves staged + all done reading buffer b^1
    if (ks + 1 < NK) stage((ks + 1) * 32, b ^ 1);  // overlap with compute

    FragAB fa, fg0, fu0, fg1, fu1;
    loadFragA(fa, shA[b], wv * 16 + ln, hi);
    loadFragB(fg0, shG[b], ln, hi);
    loadFragB(fu0, shU[b], ln, hi);
    loadFragB(fg1, shG[b], 16 + ln, hi);
    loadFragB(fu1, shU[b], 16 + ln, hi);

    accG[0] = __builtin_amdgcn_wmma_f32_16x16x32_bf16(
        false, fa.v, false, fg0.v, (short)0, accG[0], false, false);
    accU[0] = __builtin_amdgcn_wmma_f32_16x16x32_bf16(
        false, fa.v, false, fu0.v, (short)0, accU[0], false, false);
    accG[1] = __builtin_amdgcn_wmma_f32_16x16x32_bf16(
        false, fa.v, false, fg1.v, (short)0, accG[1], false, false);
    accU[1] = __builtin_amdgcn_wmma_f32_16x16x32_bf16(
        false, fa.v, false, fu1.v, (short)0, accU[1], false, false);
  }

  // Epilogue: a = u * silu(g), bf16 store. C/D layout: N=ln, M=r+8*hi.
#pragma unroll
  for (int ns = 0; ns < 2; ++ns) {
#pragma unroll
    for (int r = 0; r < 8; ++r) {
      int row = m0 + wv * 16 + r + hi * 8;
      if (row < cnt) {
        float g = accG[ns][r];
        float u = accU[ns][r];
        float a = u * (g * sigmoidf(g));
        int col = n0 + ns * 16 + ln;
        abuf[(size_t)(base + row) * I_DIM + col] = f2bf(a);
      }
    }
  }
}

// ------------------------------------------------------- GEMM2: down-proj ----
// out[t] = A @ Wd^T ; A = abuf rows [cnt, I] bf16, Wd[e] is [H, I] row-major.
__global__ __launch_bounds__(128) void k_gemm_down(
    const unsigned short* __restrict__ abuf,
    const unsigned short* __restrict__ wdb,
    const int* __restrict__ off,
    const int* __restrict__ tok_of_slot,
    float* __restrict__ out) {
  int e = blockIdx.z;
  int base = off[e];
  int cnt = off[e + 1] - base;
  int m0 = blockIdx.x * 64;
  if (m0 >= cnt) return;
  int n0 = blockIdx.y * 32;

  __shared__ unsigned short shA[2][64 * SAP];
  __shared__ unsigned short shB[2][32 * SAP];

  int tid = threadIdx.x;
  int lane = tid & 31, wv = tid >> 5;
  int hi = lane >> 4, ln = lane & 15;

  const v8f vzero = {0.f, 0.f, 0.f, 0.f, 0.f, 0.f, 0.f, 0.f};
  v8f acc[2] = {vzero, vzero};

  const unsigned short* aA  = abuf + (size_t)base * I_DIM;
  const unsigned short* wdE = wdb  + (size_t)e * H_DIM * I_DIM;

  int rA0 = (tid)       >> 2, cA0 = ((tid)       & 3) * 8;
  int rA1 = (tid + 128) >> 2, cA1 = ((tid + 128) & 3) * 8;
  int rB  = tid >> 2,         cB  = (tid & 3) * 8;

  auto stage = [&](int k0, int b) {
    async_ld16(&shA[b][rA0 * SAP + cA0],
               aA + (size_t)(m0 + rA0) * I_DIM + k0 + cA0);
    async_ld16(&shA[b][rA1 * SAP + cA1],
               aA + (size_t)(m0 + rA1) * I_DIM + k0 + cA1);
    async_ld16(&shB[b][rB * SAP + cB],
               wdE + (size_t)(n0 + rB) * I_DIM + k0 + cB);
  };

  const int NK = I_DIM / 32;
  stage(0, 0);
  for (int ks = 0; ks < NK; ++ks) {
    int b = ks & 1;
    wait_async0();
    __syncthreads();
    if (ks + 1 < NK) stage((ks + 1) * 32, b ^ 1);

    FragAB fa, fb0, fb1;
    loadFragA(fa, shA[b], wv * 16 + ln, hi);
    loadFragB(fb0, shB[b], ln, hi);
    loadFragB(fb1, shB[b], 16 + ln, hi);

    acc[0] = __builtin_amdgcn_wmma_f32_16x16x32_bf16(
        false, fa.v, false, fb0.v, (short)0, acc[0], false, false);
    acc[1] = __builtin_amdgcn_wmma_f32_16x16x32_bf16(
        false, fa.v, false, fb1.v, (short)0, acc[1], false, false);
  }

  // Scatter f32 rows back to original token positions.
#pragma unroll
  for (int r = 0; r < 8; ++r) {
    int row = m0 + wv * 16 + r + hi * 8;
    if (row < cnt) {
      int t = tok_of_slot[base + row];
#pragma unroll
      for (int ns = 0; ns < 2; ++ns) {
        int col = n0 + ns * 16 + ln;
        out[(size_t)t * H_DIM + col] = acc[ns][r];
      }
    }
  }
}

// ---------------------------------------------------------------- launch -----
extern "C" void kernel_launch(void* const* d_in, const int* in_sizes, int n_in,
                              void* d_out, int out_size, void* d_ws, size_t ws_size,
                              hipStream_t stream) {
  const float* x  = (const float*)d_in[0];  // [T, H]
  const float* gw = (const float*)d_in[1];  // [E, H]
  const float* wg = (const float*)d_in[2];  // [E, I, H]
  const float* wu = (const float*)d_in[3];  // [E, I, H]
  const float* wd = (const float*)d_in[4];  // [E, H, I]
  float* out = (float*)d_out;               // [T, H]
  (void)in_sizes; (void)n_in; (void)out_size; (void)ws_size;

  char* ws = (char*)d_ws;
  size_t p = 0;
  auto carve = [&](size_t bytes) -> char* {
    char* ptr = ws + p;
    p = (p + bytes + 255) & ~(size_t)255;
    return ptr;
  };
  int*   cnt = (int*)carve(E_NUM * sizeof(int));
  int*   off = (int*)carve((E_NUM + 1) * sizeof(int));
  int*   cur = (int*)carve(E_NUM * sizeof(int));
  int*   sel = (int*)carve((size_t)T_TOK * sizeof(int));
  float* rw  = (float*)carve((size_t)T_TOK * sizeof(float));
  int*   tok = (int*)carve((size_t)T_TOK * sizeof(int));
  unsigned short* xg   = (unsigned short*)carve((size_t)T_TOK * H_DIM * 2);
  unsigned short* abuf = (unsigned short*)carve((size_t)T_TOK * I_DIM * 2);
  unsigned short* wgb  = (unsigned short*)carve((size_t)E_NUM * I_DIM * H_DIM * 2);
  unsigned short* wub  = (unsigned short*)carve((size_t)E_NUM * I_DIM * H_DIM * 2);
  unsigned short* wdb  = (unsigned short*)carve((size_t)E_NUM * H_DIM * I_DIM * 2);

  k_zero<<<1, 64, 0, stream>>>(cnt);
  k_route<<<T_TOK / 8, 256, 0, stream>>>(x, gw, sel, rw, cnt);
  k_scan<<<1, 32, 0, stream>>>(cnt, off, cur);
  k_assign<<<T_TOK / 256, 256, 0, stream>>>(sel, cur, tok);

  const size_t Nw = (size_t)E_NUM * I_DIM * H_DIM;       // 4194304 per tensor
  k_cvt_w<<<(unsigned)((3 * Nw / 4) / 256), 256, 0, stream>>>(wg, wu, wd, wgb, wub, wdb);
  k_fill_xg<<<T_TOK, 256, 0, stream>>>(x, rw, tok, xg);

  k_gemm_gu<<<dim3(T_TOK / 64, I_DIM / 32, E_NUM), 128, 0, stream>>>(
      xg, wgb, wub, off, abuf);
  k_gemm_down<<<dim3(T_TOK / 64, H_DIM / 32, E_NUM), 128, 0, stream>>>(
      abuf, wdb, off, tok, out);
}
